// ClusterML_11742440587416
// MI455X (gfx1250) — compile-verified, matrix operand order checked by reading, not run on hardware
//
#include <hip/hip_runtime.h>
#include <hip/hip_bf16.h>

// Problem dimensions (fixed by setup_inputs in the reference)
#define B_N 131072
#define I_N 512
#define E_N 128
#define C_N 64

typedef __bf16 v16bf __attribute__((ext_vector_type(16)));
typedef float  v8f   __attribute__((ext_vector_type(8)));

#define WMMA_BF16(a, b, c) \
  __builtin_amdgcn_wmma_f32_16x16x32_bf16(false, (a), false, (b), (short)0, (c), false, false)

// -----------------------------------------------------------------------------
// Load a 16x32 bf16 operand tile (A or B of an NT GEMM) from a row-major f32
// matrix with leading dimension ldk, in the CDNA5 ISA 7.12.2 16-bit operand
// layout: lanes 0-15 hold rows 0-15 with K = k0+0..7 in elements 0..7 and
// K = k0+16..23 in elements 8..15; lanes 16-31 hold the complementary ranges.
// -----------------------------------------------------------------------------
__device__ __forceinline__ v16bf load_tile_bf16(const float* __restrict__ base,
                                                int ldk, int k0, int lane) {
  const int row = lane & 15;
  const int hh  = lane >> 4;
  const float* p = base + (size_t)row * ldk + k0 + 8 * hh;
  v16bf r;
#pragma unroll
  for (int i = 0; i < 8; ++i) r[i] = (__bf16)p[i];
#pragma unroll
  for (int i = 0; i < 8; ++i) r[8 + i] = (__bf16)p[16 + i];
  return r;
}

// -----------------------------------------------------------------------------
// Kernel 0: zero segment accumulators (re-zeroed every launch: determinism).
// -----------------------------------------------------------------------------
__global__ void zero_acc_kernel(float* __restrict__ gsum, float* __restrict__ gcnt) {
  for (int i = threadIdx.x; i < C_N * E_N; i += blockDim.x) gsum[i] = 0.0f;
  if (threadIdx.x < C_N) gcnt[threadIdx.x] = 0.0f;
}

// -----------------------------------------------------------------------------
// Kernel 1: one-time conversion of W [E,I] f32 -> swizzled bf16 WMMA operand
// chunks: wbf[(nt*16 + s)*32 + lane] = 32B chunk for column tile nt, K-step s.
// grid = 8 blocks (one per column tile) x 256 threads (8 waves, 2 steps each).
// -----------------------------------------------------------------------------
__global__ __launch_bounds__(256) void wconv_kernel(const float* __restrict__ W,
                                                    v16bf* __restrict__ wbf) {
  const int nt   = blockIdx.x;          // column tile 0..7
  const int wave = threadIdx.x >> 5;
  const int lane = threadIdx.x & 31;
  const float* base = W + (size_t)nt * 16 * I_N;
#pragma unroll
  for (int i = 0; i < 2; ++i) {
    const int s = wave + 8 * i;         // K-step 0..15
    wbf[((size_t)nt * 16 + s) * 32 + lane] = load_tile_bf16(base, I_N, 32 * s, lane);
  }
}

// -----------------------------------------------------------------------------
// Kernel 2: out[B,E] = x @ W^T + b. Block = 64 rows x 128 cols, 8 waves.
// Stage A (64x512 f32 -> bf16, swizzled) in 64 KB LDS once; wave nt owns
// column tile nt with 4 independent accumulators (hazard distance >= 4).
// B operand is the pre-swizzled bf16 W (perfectly coalesced 32 B/lane loads).
// -----------------------------------------------------------------------------
__global__ __launch_bounds__(256) void gemm_bf16_kernel(const float* __restrict__ x,
                                                        const v16bf* __restrict__ wbf,
                                                        const float* __restrict__ bias,
                                                        float* __restrict__ out) {
  __shared__ v16bf sA[4 * 16 * 32];     // [rt][s][lane] -> 64 KB
  const int m0   = blockIdx.x * 64;
  const int wave = threadIdx.x >> 5;
  const int lane = threadIdx.x & 31;

  // cooperative A staging: 64 (rt,s) chunks over 8 waves
#pragma unroll
  for (int i = 0; i < 8; ++i) {
    const int o  = wave + 8 * i;        // 0..63
    const int rt = o >> 4;
    const int s  = o & 15;
    const float* base = x + (size_t)(m0 + rt * 16) * I_N;
    __builtin_prefetch(base + (size_t)(lane & 15) * I_N + 32 * s + 8, 0, 1);
    sA[(size_t)o * 32 + lane] = load_tile_bf16(base, I_N, 32 * s, lane);
  }
  __syncthreads();

  const int nt = wave;                  // column tile
  const v16bf* wb = wbf + (size_t)nt * 16 * 32;
  v8f acc0 = {}, acc1 = {}, acc2 = {}, acc3 = {};
#pragma unroll 4
  for (int s = 0; s < 16; ++s) {
    const v16bf b  = wb[s * 32 + lane];
    const v16bf a0 = sA[(0 * 16 + s) * 32 + lane];
    const v16bf a1 = sA[(1 * 16 + s) * 32 + lane];
    const v16bf a2 = sA[(2 * 16 + s) * 32 + lane];
    const v16bf a3 = sA[(3 * 16 + s) * 32 + lane];
    acc0 = WMMA_BF16(a0, b, acc0);
    acc1 = WMMA_BF16(a1, b, acc1);
    acc2 = WMMA_BF16(a2, b, acc2);
    acc3 = WMMA_BF16(a3, b, acc3);
  }

  // D layout: lanes 0-15 -> M=j, N=lane; lanes 16-31 -> M=8+j, N=lane-16
  const int row = lane & 15;
  const int hh  = lane >> 4;
  const int n   = nt * 16 + row;
  const float bv = bias[n];
  const v8f accs[4] = {acc0, acc1, acc2, acc3};
#pragma unroll
  for (int rt = 0; rt < 4; ++rt) {
#pragma unroll
    for (int j = 0; j < 8; ++j) {
      const int m = m0 + rt * 16 + j + 8 * hh;
      out[(size_t)m * E_N + n] = accs[rt][j] + bv;
    }
  }
}

// -----------------------------------------------------------------------------
// Kernel 3: segment sums per class via 32 KB LDS accumulator (ds_add_f32),
// flushed with one global atomicAdd per (class, e). grid = B/512 blocks.
// -----------------------------------------------------------------------------
__global__ __launch_bounds__(256) void segsum_kernel(const float* __restrict__ out,
                                                     const int* __restrict__ y,
                                                     float* __restrict__ gsum,
                                                     float* __restrict__ gcnt) {
  __shared__ float lsum[C_N * E_N];
  __shared__ float lcnt[C_N];
  const int tid  = threadIdx.x;
  const int warp = tid >> 5;
  const int lane = tid & 31;

  for (int i = tid; i < C_N * E_N; i += 256) lsum[i] = 0.0f;
  if (tid < C_N) lcnt[tid] = 0.0f;
  __syncthreads();

  const int r0 = blockIdx.x * 512;
  for (int r = r0 + warp; r < r0 + 512; r += 8) {
    const int cls = y[r];
    const float* row = out + (size_t)r * E_N;
#pragma unroll
    for (int i = 0; i < 4; ++i) {
      const int e = lane + 32 * i;
      atomicAdd(&lsum[cls * E_N + e], row[e]);
    }
    if (lane == 0) atomicAdd(&lcnt[cls], 1.0f);
  }
  __syncthreads();

  for (int i = tid; i < C_N * E_N; i += 256) atomicAdd(&gsum[i], lsum[i]);
  if (tid < C_N) atomicAdd(&gcnt[tid], lcnt[tid]);
}

// -----------------------------------------------------------------------------
// Kernel 4 (single block): centroids = centroid_in + sums/counts, squared
// norms, proto_dist = exp(-0.5 * dist(centroid, centroid)) -> d_out tail.
// -----------------------------------------------------------------------------
__global__ __launch_bounds__(256) void centroid_proto_kernel(const float* __restrict__ cent_in,
                                                             const float* __restrict__ gsum,
                                                             const float* __restrict__ gcnt,
                                                             float* __restrict__ cent_out,
                                                             float* __restrict__ cnorm,
                                                             float* __restrict__ proto_out) {
  __shared__ float sc[C_N * E_N];
  __shared__ float scn[C_N];
  const int tid = threadIdx.x;

  for (int i = tid; i < C_N * E_N; i += 256) {
    const int c = i >> 7;  // / E_N
    const float v = cent_in[i] + gsum[i] / gcnt[c];
    sc[i] = v;
    cent_out[i] = v;
  }
  __syncthreads();

  if (tid < C_N) {
    float s = 0.0f;
    for (int e = 0; e < E_N; ++e) { const float v = sc[tid * E_N + e]; s += v * v; }
    scn[tid] = s;
    cnorm[tid] = s;
  }
  __syncthreads();

  for (int p = tid; p < C_N * C_N; p += 256) {
    const int i = p >> 6;
    const int j = p & 63;
    float dot = 0.0f;
    for (int e = 0; e < E_N; ++e) dot += sc[i * E_N + e] * sc[j * E_N + e];
    float d2 = scn[i] + scn[j] - 2.0f * dot;
    d2 = fmaxf(d2, 1e-12f);
    proto_out[p] = expf(-0.5f * sqrtf(d2));
  }
}

// -----------------------------------------------------------------------------
// Kernel 5: result = sigmoid(exp(-0.5 * ||out_r - cent_c||)).
// Block = 32 rows x 64 cols; 8 waves = 2 row tiles x 4 col tiles.
// out rows and centroid staged once in LDS as swizzled bf16; exact f32 row
// norms accumulated during staging (shfl_xor(16) pairs the K half-lanes, then
// one ds_add_f32 per (row, K-chunk)). Inner loop: ds_load + wmma, dual accum.
// -----------------------------------------------------------------------------
__global__ __launch_bounds__(256) void dist_sig_kernel(const float* __restrict__ out,
                                                       const float* __restrict__ cent,
                                                       const float* __restrict__ cnorm,
                                                       float* __restrict__ res) {
  __shared__ v16bf sA[8 * 32];    // [rt*4+s][lane]  (2 rt x 4 s)  -> 8 KB
  __shared__ v16bf sB[16 * 32];   // [ct*4+s][lane]  (4 ct x 4 s)  -> 16 KB
  __shared__ float srn[32];
  __shared__ float scn[C_N];

  const int tid  = threadIdx.x;
  const int wave = tid >> 5;
  const int lane = tid & 31;
  const int m0   = blockIdx.x * 32;
  const int row  = lane & 15;
  const int hh   = lane >> 4;

  if (tid < 32) srn[tid] = 0.0f;
  if (tid < C_N) scn[tid] = cnorm[tid];
  __syncthreads();

  // stage A (this block's 32 out-rows) + accumulate exact f32 row norms
  {
    const int art = wave >> 2;          // 0..1
    const int as  = wave & 3;           // K-step 0..3
    const float* p = out + (size_t)(m0 + art * 16 + row) * E_N + 32 * as + 8 * hh;
    v16bf a;
    float ss = 0.0f;
#pragma unroll
    for (int i = 0; i < 8; ++i) { const float v = p[i];      ss += v * v; a[i]     = (__bf16)v; }
#pragma unroll
    for (int i = 0; i < 8; ++i) { const float v = p[16 + i]; ss += v * v; a[8 + i] = (__bf16)v; }
    sA[wave * 32 + lane] = a;
    ss += __shfl_xor(ss, 16);           // merge the two K half-lanes of this row
    if (hh == 0) atomicAdd(&srn[art * 16 + row], ss);
  }

  // stage B (centroids, 64x128): 16 chunks over 8 waves
#pragma unroll
  for (int i = 0; i < 2; ++i) {
    const int o  = wave * 2 + i;        // 0..15
    const int ct = o >> 2;
    const int s  = o & 3;
    sB[o * 32 + lane] =
        load_tile_bf16(cent + (size_t)(ct * 16) * E_N, E_N, 32 * s, lane);
  }
  __syncthreads();

  const int rt = wave >> 2;             // row tile 0..1
  const int ct = wave & 3;              // col tile 0..3
  v8f acc0 = {}, acc1 = {};
  acc0 = WMMA_BF16(sA[(rt * 4 + 0) * 32 + lane], sB[(ct * 4 + 0) * 32 + lane], acc0);
  acc1 = WMMA_BF16(sA[(rt * 4 + 1) * 32 + lane], sB[(ct * 4 + 1) * 32 + lane], acc1);
  acc0 = WMMA_BF16(sA[(rt * 4 + 2) * 32 + lane], sB[(ct * 4 + 2) * 32 + lane], acc0);
  acc1 = WMMA_BF16(sA[(rt * 4 + 3) * 32 + lane], sB[(ct * 4 + 3) * 32 + lane], acc1);

  const int n = ct * 16 + row;
  const float cn = scn[n];
#pragma unroll
  for (int j = 0; j < 8; ++j) {
    const int mloc = rt * 16 + j + 8 * hh;
    float d2 = srn[mloc] + cn - 2.0f * (acc0[j] + acc1[j]);
    d2 = fmaxf(d2, 1e-12f);
    const float t = expf(-0.5f * sqrtf(d2));
    res[(size_t)(m0 + mloc) * C_N + n] = 1.0f / (1.0f + expf(-t));
  }
}

// -----------------------------------------------------------------------------
// Host-side launcher
// -----------------------------------------------------------------------------
extern "C" void kernel_launch(void* const* d_in, const int* in_sizes, int n_in,
                              void* d_out, int out_size, void* d_ws, size_t ws_size,
                              hipStream_t stream) {
  (void)in_sizes; (void)n_in; (void)out_size; (void)ws_size;

  const float* x        = (const float*)d_in[0];   // [B, I]
  const int*   y        = (const int*)d_in[1];     // [B]
  const float* W        = (const float*)d_in[2];   // [E, I]
  const float* b        = (const float*)d_in[3];   // [E]
  const float* centroid = (const float*)d_in[4];   // [C, E]

  float* outp = (float*)d_out;                     // [B*C] then [C*C]

  // workspace layout
  float* ws       = (float*)d_ws;
  float* ws_out   = ws;                                  // B*E f32
  float* ws_sum   = ws_out + (size_t)B_N * E_N;          // C*E f32
  float* ws_cnt   = ws_sum + (size_t)C_N * E_N;          // C   f32
  float* ws_cent  = ws_cnt + C_N;                        // C*E f32
  float* ws_cnorm = ws_cent + (size_t)C_N * E_N;         // C   f32
  v16bf* ws_wbf   = (v16bf*)(ws_cnorm + C_N);            // 8*16*32 chunks (128 KB), 32B-aligned

  zero_acc_kernel<<<1, 256, 0, stream>>>(ws_sum, ws_cnt);

  wconv_kernel<<<E_N / 16, 256, 0, stream>>>(W, ws_wbf);

  gemm_bf16_kernel<<<B_N / 64, 256, 0, stream>>>(x, ws_wbf, b, ws_out);

  segsum_kernel<<<B_N / 512, 256, 0, stream>>>(ws_out, y, ws_sum, ws_cnt);

  centroid_proto_kernel<<<1, 256, 0, stream>>>(centroid, ws_sum, ws_cnt,
                                               ws_cent, ws_cnorm,
                                               outp + (size_t)B_N * C_N);

  dist_sig_kernel<<<B_N / 32, 256, 0, stream>>>(ws_out, ws_cent, ws_cnorm, outp);
}
